// AIMNet2_24816321036387
// MI455X (gfx1250) — matrix operand order repaired and checked by reading, not compiled, not run on hardware
//
#include <hip/hip_runtime.h>
#include <hip/hip_bf16.h>
#include <math.h>

// ---------------------------------------------------------------------------
// AIMNet2 forward on gfx1250 (MI455X). All dense contractions run through
// v_wmma_f32_16x16x32_f16 (f16 inputs, f32 accumulate). Problem is tiny
// (~9 GFLOP, ~100MB traffic, everything L2-resident) => latency-bound.
// Round 2: operand feeds vectorized -- A tiles as 2x b128 loads from
// row-major f16, B operands pre-swizzled on device into the WMMA B register
// layout (one 32B record per lane per tile -> single v16h load).
// ---------------------------------------------------------------------------

#define BB 8
#define NN 192
#define GG 16
#define FF 16
#define HSS 16
#define HVV 8
#define ROWS (BB * NN)      // 1536
#define XW 672              // padded MLP input width (640 / 665 -> 672 = 21*32)
#define OW 272              // padded MLP output width (258 -> 272 = 17*16)

typedef __attribute__((ext_vector_type(16))) _Float16 v16h;
typedef __attribute__((ext_vector_type(8)))  _Float16 v8h;
typedef __attribute__((ext_vector_type(8)))  float    v8f;

// ---- WMMA operand loaders --------------------------------------------------
// A: 16(M) x 32(K) from row-major f16 A[row*lda + k]; this code gives each
// lane contiguous K=[kb..kb+15] (kb = k0 + 16*(lane/16)), same order as B packing -> two 16B loads.
// Requires lda % 8 == 0 (16B row alignment) -- true for all our matrices.
__device__ __forceinline__ v16h load_a_tile(const _Float16* __restrict__ A,
                                            int lda, int row0, int k0, int lane) {
  const int r  = row0 + (lane & 15);
  const int kb = k0 + ((lane & 16) ? 16 : 0);
  const _Float16* p = A + (size_t)r * lda + kb;
  const v8h lo = *(const v8h*)p;
  const v8h hi = *(const v8h*)(p + 8);
  return __builtin_shufflevector(lo, hi, 0, 1, 2, 3, 4, 5, 6, 7,
                                 8, 9, 10, 11, 12, 13, 14, 15);
}

// B: pre-swizzled tile-major layout [kt][tn][lane][16 halves] (32B records).
// Record for (kt,tn,lane) holds B[kt*32 + (lane&16?16:0) + i][tn*16 + (lane&15)].
__device__ __forceinline__ v16h load_b_sw(const _Float16* __restrict__ Bsw,
                                          int tiles_n, int kt, int tn, int lane) {
  const _Float16* p = Bsw + ((size_t)(kt * tiles_n + tn) * 32 + lane) * 16;
  return *(const v16h*)p;
}

__device__ __forceinline__ float gelu_exact(float v) {
  return 0.5f * v * (1.0f + erff(v * 0.70710678118654752f));
}

// ---- small utility kernels -------------------------------------------------
__global__ void k_fill0(float* p, int n) {
  int t = blockIdx.x * blockDim.x + threadIdx.x;
  if (t < n) p[t] = 0.0f;
}

__global__ void k_f32_to_f16(const float* __restrict__ s, _Float16* __restrict__ d, int n) {
  int t = blockIdx.x * blockDim.x + threadIdx.x;
  if (t < n) d[t] = (_Float16)s[t];
}

// f32 W [K x Nsrc] row-major -> f16 swizzled B layout [Kpad/32][Npad/16][32][16]
__global__ void k_pack_w_sw(const float* __restrict__ W, _Float16* __restrict__ Wh,
                            int K, int Nsrc, int Kpad, int Npad) {
  int t = blockIdx.x * blockDim.x + threadIdx.x;
  if (t >= Kpad * Npad) return;
  const int i    = t & 15;
  const int rec  = t >> 4;
  const int lane = rec & 31;
  const int tile = rec >> 5;
  const int tiles_n = Npad >> 4;
  const int kt = tile / tiles_n, tn = tile % tiles_n;
  const int k = kt * 32 + ((lane & 16) ? 16 : 0) + i;
  const int n = tn * 16 + (lane & 15);
  Wh[t] = (k < K && n < Nsrc) ? (_Float16)W[(size_t)k * Nsrc + n] : (_Float16)0.0f;
}

__global__ void k_pack_b(const float* __restrict__ b, float* __restrict__ bp,
                         int Nsrc, int Npad) {
  int t = blockIdx.x * blockDim.x + threadIdx.x;
  if (t < Npad) bp[t] = (t < Nsrc) ? b[t] : 0.0f;
}

// ---- AEV: gs_h [b][g][n][m], gv_h [b][d][g][n][m] (f16, A-operand role) ----
__global__ void k_aev(const float* __restrict__ coord, const int* __restrict__ numbers,
                      _Float16* __restrict__ gs_h, _Float16* __restrict__ gv_h) {
  int t = blockIdx.x * blockDim.x + threadIdx.x;
  if (t >= BB * NN * NN) return;
  const int m = t % NN, n = (t / NN) % NN, b = t / (NN * NN);
  const float* cn = coord + (size_t)(b * NN + n) * 3;
  const float* cm = coord + (size_t)(b * NN + m) * 3;
  const float rx = cn[0] - cm[0], ry = cn[1] - cm[1], rz = cn[2] - cm[2];
  const float d2 = rx * rx + ry * ry + rz * rz;
  const float d = sqrtf(d2 > 0.0f ? d2 : 1.0f);
  const bool valid = (n != m) && (numbers[b * NN + m] != 0) && (d < 5.0f);
  const float inv = 1.0f / d;
  const float ux = rx * inv, uy = ry * inv, uz = rz * inv;
  float tt = d * 0.2f; if (tt > 1.0f) tt = 1.0f;
  const float fc = 0.5f * cosf(3.14159265358979f * tt) + 0.5f;
  const float eta = (16.0f / 4.2f) * (16.0f / 4.2f);
  const size_t nm = (size_t)n * NN + m;
  const size_t plane = (size_t)NN * NN;
#pragma unroll
  for (int g = 0; g < GG; ++g) {
    const float sh = 0.8f + 4.2f * (float)g / 16.0f;
    const float dd = d - sh;
    const float gval = valid ? expf(-eta * dd * dd) * fc : 0.0f;
    gs_h[((size_t)(b * GG + g)) * plane + nm] = (_Float16)gval;
    gv_h[((size_t)((b * 3 + 0) * GG + g)) * plane + nm] = (_Float16)(ux * gval);
    gv_h[((size_t)((b * 3 + 1) * GG + g)) * plane + nm] = (_Float16)(uy * gval);
    gv_h[((size_t)((b * 3 + 2) * GG + g)) * plane + nm] = (_Float16)(uz * gval);
  }
}

// ---- a init / pack ---------------------------------------------------------
__global__ void k_init_a(const float* __restrict__ afv, const int* __restrict__ numbers,
                         float* __restrict__ a) {
  int t = blockIdx.x * blockDim.x + threadIdx.x;
  if (t >= ROWS * FF * GG) return;
  int row = t / (FF * GG), c = t % (FF * GG);
  a[t] = afv[(size_t)numbers[row] * (FF * GG) + c];
}

// a [row][f][g] (f32) -> per-(b,g) swizzled B operand (192x16), 6 k-tiles.
// a_h layout: [bg][kt(6)][lane(32)][i(16)]
__global__ void k_pack_a_sw(const float* __restrict__ a, _Float16* __restrict__ a_h) {
  int t = blockIdx.x * blockDim.x + threadIdx.x;
  if (t >= BB * GG * NN * FF) return;
  const int i    = t & 15;
  const int rec  = t >> 4;
  const int lane = rec & 31;
  const int kt   = (rec >> 5) % 6;
  const int bg   = rec / (32 * 6) / 1;  // rec >> 5 / 6
  const int b = bg / GG, g = bg % GG;
  const int k = kt * 32 + ((lane & 16) ? 16 : 0) + i;   // neighbor index m
  const int f = lane & 15;
  a_h[t] = (_Float16)a[((size_t)(b * NN + k) * FF + f) * GG + g];
}

// ---- conv GEMM: per (b,g,ntile) wave; C = gs/gv[192x192] @ a[192x16] -------
__global__ void k_conv_wmma(const _Float16* __restrict__ gs_h,
                            const _Float16* __restrict__ gv_h,
                            const _Float16* __restrict__ a_h,
                            float* __restrict__ s1, float* __restrict__ v1) {
  const int NT = NN / 16; // 12
  const int wave = blockIdx.x * (blockDim.x >> 5) + (threadIdx.x >> 5);
  if (wave >= BB * GG * NT) return;
  const int tn = wave % NT, bg = wave / NT;
  const int b = bg / GG, g = bg % GG;
  const int lane = threadIdx.x & 31;
  const size_t plane = (size_t)NN * NN;
  const _Float16* As  = gs_h + (size_t)bg * plane;
  const _Float16* Av0 = gv_h + (size_t)((b * 3 + 0) * GG + g) * plane;
  const _Float16* Av1 = gv_h + (size_t)((b * 3 + 1) * GG + g) * plane;
  const _Float16* Av2 = gv_h + (size_t)((b * 3 + 2) * GG + g) * plane;
  const _Float16* Bm  = a_h + (size_t)bg * NN * FF;   // swizzled, tiles_n = 1
  v8f accs = {}, acc0 = {}, acc1 = {}, acc2 = {};
  for (int kt = 0; kt < NN / 32; ++kt) {
    const v16h bt = load_b_sw(Bm, 1, kt, 0, lane);    // reused by 4 A sources
    v16h at = load_a_tile(As, NN, tn * 16, kt * 32, lane);
    accs = __builtin_amdgcn_wmma_f32_16x16x32_f16(false, at, false, bt, (short)0, accs, false, false);
    at = load_a_tile(Av0, NN, tn * 16, kt * 32, lane);
    acc0 = __builtin_amdgcn_wmma_f32_16x16x32_f16(false, at, false, bt, (short)0, acc0, false, false);
    at = load_a_tile(Av1, NN, tn * 16, kt * 32, lane);
    acc1 = __builtin_amdgcn_wmma_f32_16x16x32_f16(false, at, false, bt, (short)0, acc1, false, false);
    at = load_a_tile(Av2, NN, tn * 16, kt * 32, lane);
    acc2 = __builtin_amdgcn_wmma_f32_16x16x32_f16(false, at, false, bt, (short)0, acc2, false, false);
  }
  const int f = lane & 15;
  const int r0 = tn * 16 + ((lane & 16) ? 8 : 0);
#pragma unroll
  for (int r = 0; r < 8; ++r) {
    const size_t row = (size_t)b * NN + (r0 + r);
    s1[(row * FF + f) * GG + g] = accs[r];
    v1[((row * 3 + 0) * FF + f) * GG + g] = acc0[r];
    v1[((row * 3 + 1) * FF + f) * GG + g] = acc1[r];
    v1[((row * 3 + 2) * FF + f) * GG + g] = acc2[r];
  }
}

// ---- conv mix: contract g with Ws/Wv, vector norm; fill x[0..640) ----------
__global__ void k_conv_mix(const float* __restrict__ a, const float* __restrict__ s1,
                           const float* __restrict__ v1, const float* __restrict__ Ws,
                           const float* __restrict__ Wv, float* __restrict__ x) {
  int t = blockIdx.x * blockDim.x + threadIdx.x;
  if (t >= ROWS * FF) return;
  const int f = t % FF, row = t / FF;
  float* xr = x + (size_t)row * XW;
  float sg[GG], vg0[GG], vg1[GG], vg2[GG];
#pragma unroll
  for (int g = 0; g < GG; ++g) {
    sg[g]  = s1[((size_t)row * FF + f) * GG + g];
    vg0[g] = v1[(((size_t)row * 3 + 0) * FF + f) * GG + g];
    vg1[g] = v1[(((size_t)row * 3 + 1) * FF + f) * GG + g];
    vg2[g] = v1[(((size_t)row * 3 + 2) * FF + f) * GG + g];
    xr[f * GG + g] = a[((size_t)row * FF + f) * GG + g]; // copy features
  }
#pragma unroll
  for (int h = 0; h < HSS; ++h) {
    float acc = 0.0f;
#pragma unroll
    for (int g = 0; g < GG; ++g) acc += sg[g] * Ws[(f * GG + g) * HSS + h];
    xr[256 + f * HSS + h] = acc;
  }
#pragma unroll
  for (int h = 0; h < HVV; ++h) {
    float a0 = 0.0f, a1 = 0.0f, a2 = 0.0f;
#pragma unroll
    for (int g = 0; g < GG; ++g) {
      const float w = Wv[(f * GG + g) * HVV + h];
      a0 += vg0[g] * w; a1 += vg1[g] * w; a2 += vg2[g] * w;
    }
    xr[512 + f * HVV + h] = sqrtf(a0 * a0 + a1 * a1 + a2 * a2 + 1e-12f);
  }
}

// ---- conv_q: neighbor-weighted charge sums ---------------------------------
__global__ void k_convq_dot(const _Float16* __restrict__ gs_h, const _Float16* __restrict__ gv_h,
                            const float* __restrict__ q, float* __restrict__ tq,
                            float* __restrict__ tqv) {
  int t = blockIdx.x * blockDim.x + threadIdx.x;
  if (t >= ROWS * GG) return;
  const int g = t % GG, n = (t / GG) % NN, b = t / (GG * NN);
  const size_t plane = (size_t)NN * NN;
  const _Float16* gsr = gs_h + ((size_t)(b * GG + g)) * plane + (size_t)n * NN;
  const _Float16* g0  = gv_h + ((size_t)((b * 3 + 0) * GG + g)) * plane + (size_t)n * NN;
  const _Float16* g1  = gv_h + ((size_t)((b * 3 + 1) * GG + g)) * plane + (size_t)n * NN;
  const _Float16* g2  = gv_h + ((size_t)((b * 3 + 2) * GG + g)) * plane + (size_t)n * NN;
  float as = 0.0f, a0 = 0.0f, a1 = 0.0f, a2 = 0.0f;
  for (int m = 0; m < NN; ++m) {
    const float qm = q[b * NN + m];
    as += qm * (float)gsr[m];
    a0 += qm * (float)g0[m];
    a1 += qm * (float)g1[m];
    a2 += qm * (float)g2[m];
  }
  const size_t row = (size_t)b * NN + n;
  tq[row * GG + g] = as;
  tqv[(row * 3 + 0) * GG + g] = a0;
  tqv[(row * 3 + 1) * GG + g] = a1;
  tqv[(row * 3 + 2) * GG + g] = a2;
}

__global__ void k_convq_mix(const float* __restrict__ q, const float* __restrict__ tq,
                            const float* __restrict__ tqv, const float* __restrict__ cqs,
                            const float* __restrict__ cqv, float* __restrict__ x) {
  int row = blockIdx.x * blockDim.x + threadIdx.x;
  if (row >= ROWS) return;
  float* xr = x + (size_t)row * XW;
  xr[640] = q[row];
  float tg[GG], t0[GG], t1[GG], t2[GG];
#pragma unroll
  for (int g = 0; g < GG; ++g) {
    tg[g] = tq[(size_t)row * GG + g];
    t0[g] = tqv[((size_t)row * 3 + 0) * GG + g];
    t1[g] = tqv[((size_t)row * 3 + 1) * GG + g];
    t2[g] = tqv[((size_t)row * 3 + 2) * GG + g];
  }
#pragma unroll
  for (int h = 0; h < HSS; ++h) {
    float acc = 0.0f;
#pragma unroll
    for (int g = 0; g < GG; ++g) acc += tg[g] * cqs[g * HSS + h];
    xr[641 + h] = acc;
  }
#pragma unroll
  for (int h = 0; h < HVV; ++h) {
    float a0 = 0.0f, a1 = 0.0f, a2 = 0.0f;
#pragma unroll
    for (int g = 0; g < GG; ++g) {
      const float w = cqv[g * HVV + h];
      a0 += t0[g] * w; a1 += t1[g] * w; a2 += t2[g] * w;
    }
    xr[657 + h] = sqrtf(a0 * a0 + a1 * a1 + a2 * a2 + 1e-12f);
  }
}

// ---- generic WMMA GEMM + bias + (optional) exact GELU ----------------------
__global__ void k_gemm(const _Float16* __restrict__ X, const _Float16* __restrict__ W,
                       const float* __restrict__ bias, float* __restrict__ Y,
                       int M, int Np, int K, int act) {
  const int wave = blockIdx.x * (blockDim.x >> 5) + (threadIdx.x >> 5);
  const int ntn = Np >> 4;
  if (wave >= (M >> 4) * ntn) return;
  const int tm = wave / ntn, tn = wave % ntn;
  const int lane = threadIdx.x & 31;
  v8f acc = {};
  for (int kt = 0; kt < (K >> 5); ++kt) {
    v16h a = load_a_tile(X, K, tm * 16, kt * 32, lane);
    v16h b = load_b_sw(W, ntn, kt, tn, lane);
    acc = __builtin_amdgcn_wmma_f32_16x16x32_f16(false, a, false, b, (short)0, acc, false, false);
  }
  const int col = tn * 16 + (lane & 15);
  const int r0 = tm * 16 + ((lane & 16) ? 8 : 0);
  const float bv = bias[col];
#pragma unroll
  for (int r = 0; r < 8; ++r) {
    float v = acc[r] + bv;
    if (act) v = gelu_exact(v);
    Y[(size_t)(r0 + r) * Np + col] = v;
  }
}

// ---- pad-row masking, NQE, feature update ----------------------------------
__global__ void k_mask_rows(float* __restrict__ buf, const int* __restrict__ numbers, int width) {
  int t = blockIdx.x * blockDim.x + threadIdx.x;
  if (t >= ROWS * width) return;
  if (numbers[t / width] == 0) buf[t] = 0.0f;
}

__global__ void k_nqe(const float* __restrict__ ob, int ldo, const float* __restrict__ charge,
                      float* __restrict__ q, int add_prev) {
  const int b = blockIdx.x;
  const int n = threadIdx.x;
  __shared__ float sq[256], sw[256];
  float qv = 0.0f, w = 0.0f;
  if (n < NN) {
    const int row = b * NN + n;
    qv = ob[(size_t)row * ldo + 0] + (add_prev ? q[row] : 0.0f);
    const float f = ob[(size_t)row * ldo + 1];
    w = f * f;
  }
  sq[threadIdx.x] = qv; sw[threadIdx.x] = w;
  __syncthreads();
  for (int s = 128; s > 0; s >>= 1) {
    if (threadIdx.x < s) { sq[threadIdx.x] += sq[threadIdx.x + s]; sw[threadIdx.x] += sw[threadIdx.x + s]; }
    __syncthreads();
  }
  if (n < NN) {
    const int row = b * NN + n;
    q[row] = qv + (charge[b] - sq[0]) * (w / (sw[0] + 1e-8f));
  }
}

__global__ void k_a_update(float* __restrict__ a, const float* __restrict__ ob, int ldo) {
  int t = blockIdx.x * blockDim.x + threadIdx.x;
  if (t >= ROWS * FF * GG) return;
  a[t] += ob[(size_t)(t / (FF * GG)) * ldo + 2 + (t % (FF * GG))];
}

// ---------------------------------------------------------------------------
#define LT(kern, total, ...) kern<<<((total) + 255) / 256, 256, 0, stream>>>(__VA_ARGS__)

extern "C" void kernel_launch(void* const* d_in, const int* in_sizes, int n_in,
                              void* d_out, int out_size, void* d_ws, size_t ws_size,
                              hipStream_t stream) {
  (void)in_sizes; (void)n_in; (void)out_size; (void)ws_size;
  const float* coord   = (const float*)d_in[0];
  const int*   numbers = (const int*)  d_in[1];
  const float* charge  = (const float*)d_in[2];
  const float* afv     = (const float*)d_in[3];
  const float* ca_s    = (const float*)d_in[4];
  const float* ca_v    = (const float*)d_in[5];
  const float* cq_s    = (const float*)d_in[6];
  const float* cq_v    = (const float*)d_in[7];
  const float* mw[3][3]; const float* mb[3][3];
  for (int i = 0; i < 3; ++i)
    for (int l = 0; l < 3; ++l) {
      mw[i][l] = (const float*)d_in[8 + i * 6 + l * 2];
      mb[i][l] = (const float*)d_in[8 + i * 6 + l * 2 + 1];
    }
  float* out = (float*)d_out;

  // bump allocator on workspace
  char* wp = (char*)d_ws;
  auto alloc = [&](size_t bytes) -> void* {
    void* r = (void*)wp;
    wp += (bytes + 255) & ~(size_t)255;
    return r;
  };
  _Float16* gs_h = (_Float16*)alloc((size_t)BB * GG * NN * NN * 2);
  _Float16* gv_h = (_Float16*)alloc((size_t)BB * 3 * GG * NN * NN * 2);
  float*    a    = (float*)   alloc((size_t)ROWS * FF * GG * 4);
  _Float16* a_h  = (_Float16*)alloc((size_t)BB * GG * NN * FF * 2);
  float*    s1   = (float*)   alloc((size_t)ROWS * FF * GG * 4);
  float*    v1   = (float*)   alloc((size_t)ROWS * 3 * FF * GG * 4);
  float*    q    = (float*)   alloc((size_t)ROWS * 4);
  float*    tq   = (float*)   alloc((size_t)ROWS * GG * 4);
  float*    tqv  = (float*)   alloc((size_t)ROWS * 3 * GG * 4);
  float*    x    = (float*)   alloc((size_t)ROWS * XW * 4);
  _Float16* x_h  = (_Float16*)alloc((size_t)ROWS * XW * 2);
  float*    h1   = (float*)   alloc((size_t)ROWS * 512 * 4);
  _Float16* h1_h = (_Float16*)alloc((size_t)ROWS * 512 * 2);
  float*    h2   = (float*)   alloc((size_t)ROWS * 512 * 4);
  _Float16* h2_h = (_Float16*)alloc((size_t)ROWS * 512 * 2);
  float*    ob   = (float*)   alloc((size_t)ROWS * OW * 4);
  _Float16* w0h[3]; _Float16* w1h[3]; _Float16* w2h[3];
  float* b0p[3]; float* b1p[3]; float* b2p[3];
  const int K0[3]  = {640, 665, 665};   // logical input widths
  const int N2[3]  = {258, 258, 256};   // logical last-layer widths
  const int N2p[3] = {OW, OW, 256};     // padded last-layer widths
  for (int i = 0; i < 3; ++i) {
    w0h[i] = (_Float16*)alloc((size_t)XW * 512 * 2);
    w1h[i] = (_Float16*)alloc((size_t)512 * 512 * 2);
    w2h[i] = (_Float16*)alloc((size_t)512 * OW * 2);
    b0p[i] = (float*)alloc(512 * 4);
    b1p[i] = (float*)alloc(512 * 4);
    b2p[i] = (float*)alloc(OW * 4);
  }

  // pack weights/biases to f16 swizzled / padded
  for (int i = 0; i < 3; ++i) {
    LT(k_pack_w_sw, XW * 512,     mw[i][0], w0h[i], K0[i], 512,   XW,  512);
    LT(k_pack_w_sw, 512 * 512,    mw[i][1], w1h[i], 512,   512,   512, 512);
    LT(k_pack_w_sw, 512 * N2p[i], mw[i][2], w2h[i], 512,   N2[i], 512, N2p[i]);
    LT(k_pack_b, 512,    mb[i][0], b0p[i], 512,   512);
    LT(k_pack_b, 512,    mb[i][1], b1p[i], 512,   512);
    LT(k_pack_b, N2p[i], mb[i][2], b2p[i], N2[i], N2p[i]);
  }

  // AEV (one time) + initial features
  LT(k_aev, BB * NN * NN, coord, numbers, gs_h, gv_h);
  LT(k_init_a, ROWS * FF * GG, afv, numbers, a);

  auto conv_pass = [&]() {
    LT(k_fill0, ROWS * XW, x, ROWS * XW);
    LT(k_pack_a_sw, BB * GG * NN * FF, a, a_h);
    const int waves = BB * GG * (NN / 16);
    k_conv_wmma<<<(waves + 7) / 8, 256, 0, stream>>>(gs_h, gv_h, a_h, s1, v1);
    LT(k_conv_mix, ROWS * FF, a, s1, v1, ca_s, ca_v, x);
  };
  auto q_pass = [&]() {
    LT(k_convq_dot, ROWS * GG, gs_h, gv_h, q, tq, tqv);
    LT(k_convq_mix, ROWS, q, tq, tqv, cq_s, cq_v, x);
  };
  auto mlp = [&](int i, float* finalY, int finalNp, int lastAct) {
    LT(k_f32_to_f16, ROWS * XW, x, x_h, ROWS * XW);
    int t0 = (ROWS / 16) * (512 / 16);
    k_gemm<<<(t0 + 7) / 8, 256, 0, stream>>>(x_h, w0h[i], b0p[i], h1, ROWS, 512, XW, 1);
    LT(k_f32_to_f16, ROWS * 512, h1, h1_h, ROWS * 512);
    k_gemm<<<(t0 + 7) / 8, 256, 0, stream>>>(h1_h, w1h[i], b1p[i], h2, ROWS, 512, 512, 1);
    LT(k_f32_to_f16, ROWS * 512, h2, h2_h, ROWS * 512);
    int t2 = (ROWS / 16) * (finalNp / 16);
    k_gemm<<<(t2 + 7) / 8, 256, 0, stream>>>(h2_h, w2h[i], b2p[i], finalY, ROWS, finalNp, 512, lastAct);
    LT(k_mask_rows, ROWS * finalNp, finalY, numbers, finalNp);
  };

  // ---- pass 0: charges set, features delta ----
  conv_pass();
  mlp(0, ob, OW, 0);
  k_nqe<<<BB, 256, 0, stream>>>(ob, OW, charge, q, 0);
  LT(k_a_update, ROWS * FF * GG, a, ob, OW);

  // ---- pass 1: delta update (with charge conv) ----
  conv_pass();
  q_pass();
  mlp(1, ob, OW, 1);
  k_nqe<<<BB, 256, 0, stream>>>(ob, OW, charge, q, 1);
  LT(k_a_update, ROWS * FF * GG, a, ob, OW);

  // ---- pass 2: final AIM output straight into d_out ----
  conv_pass();
  q_pass();
  mlp(2, out, 256, 1);
}
